// TransformerBlock_7395933684172
// MI455X (gfx1250) — matrix operand order repaired and checked by reading, not compiled, not run on hardware
//
#include <hip/hip_runtime.h>
#include <hip/hip_bf16.h>
#include <math.h>

typedef __attribute__((ext_vector_type(16))) __bf16 v16bf;
typedef __attribute__((ext_vector_type(8)))  float  v8f;

// ---------- helpers ----------

__device__ __forceinline__ __bf16 f2bf(float f) {
  unsigned u = __builtin_bit_cast(unsigned, f);
  unsigned r = u + 0x7FFFu + ((u >> 16) & 1u);   // round-to-nearest-even
  unsigned short h = (unsigned short)(r >> 16);
  return __builtin_bit_cast(__bf16, h);
}

__device__ __forceinline__ v8f wmma_bf16(v16bf a, v16bf b, v8f c) {
  return __builtin_amdgcn_wmma_f32_16x16x32_bf16(
      /*neg_a=*/false, a, /*neg_b=*/false, b,
      /*c_mod=*/(short)0, c, /*reuse_a=*/false, /*reuse_b=*/false);
}

// A-fragment (16x32 bf16, MxK): lane = row (lane&15); K chunks of 8 at
// kb=(lane>>4)*8 and kb+16.  `base` points at row 0 of the 16-row strip.
__device__ __forceinline__ v16bf load_frag_a(const __bf16* base, int stride) {
  const int lane = threadIdx.x & 31;
  const __bf16* p = base + (lane & 15) * stride + ((lane >> 4) << 3);
  union { uint4 u[2]; v16bf v; } u;
  u.u[0] = *(const uint4*)(p);
  u.u[1] = *(const uint4*)(p + 16);
  return u.v;
}

// B-fragment (32x16 bf16, KxN) loaded from B^T layout [N][K]:
// lane = column (lane&15); K chunks of 8 at kb=(lane>>4)*8 and kb+16.
__device__ __forceinline__ v16bf load_frag_b(const __bf16* baseT, int stride) {
  const int lane = threadIdx.x & 31;
  const __bf16* p = baseT + (lane & 15) * stride + ((lane >> 4) << 3);
  union { uint4 u[2]; v16bf v; } u;
  u.u[0] = *(const uint4*)(p);
  u.u[1] = *(const uint4*)(p + 16);
  return u.v;
}

// CDNA5 async global->LDS copy (16 bytes), tracked by ASYNCcnt.
// Flat-LDS addressing uses addr[31:0] as the wave-relative LDS byte address,
// so truncating the generic pointer gives the VDST operand.
__device__ __forceinline__ unsigned lds_off(const void* p) {
  return (unsigned)(unsigned long long)p;
}
__device__ __forceinline__ void async_b128(unsigned dst_lds, const __bf16* src) {
  asm volatile("global_load_async_to_lds_b128 %0, %1, off"
               :: "v"(dst_lds), "v"(src) : "memory");
}
#define WAIT_ASYNC(n) asm volatile("s_wait_asynccnt %0" :: "n"(n) : "memory")

// ---------- weight transpose + bf16 convert: Bt[n*K+k] = B[k*N+n] ----------

__global__ void k_transpose_bf16(const float* __restrict__ B, __bf16* __restrict__ Bt,
                                 int K, int N) {
  int idx = blockIdx.x * 256 + threadIdx.x;
  int total = K * N;
  if (idx >= total) return;
  int n = idx / K, k = idx - n * K;
  Bt[idx] = f2bf(B[(size_t)k * N + n]);
}

// ---------- layernorm (1024 cols) -> bf16 ----------

__global__ __launch_bounds__(256)
void k_layernorm(const float* __restrict__ x, const float* __restrict__ g,
                 const float* __restrict__ b, __bf16* __restrict__ out) {
  __shared__ float rs[256], rs2[256];
  const int row = blockIdx.x, tid = threadIdx.x;
  const float* xr = x + (size_t)row * 1024;
  float v[4], s = 0.f, s2 = 0.f;
  for (int i = 0; i < 4; ++i) {
    v[i] = xr[tid + (i << 8)];
    s += v[i]; s2 += v[i] * v[i];
  }
  rs[tid] = s; rs2[tid] = s2;
  __syncthreads();
  for (int off = 128; off > 0; off >>= 1) {
    if (tid < off) { rs[tid] += rs[tid + off]; rs2[tid] += rs2[tid + off]; }
    __syncthreads();
  }
  const float mu = rs[0] * (1.f / 1024.f);
  const float var = rs2[0] * (1.f / 1024.f) - mu * mu;
  const float rstd = rsqrtf(var + 1e-5f);
  for (int i = 0; i < 4; ++i) {
    int c = tid + (i << 8);
    out[(size_t)row * 1024 + c] = f2bf((v[i] - mu) * rstd * g[c] + b[c]);
  }
}

// ---------- tiled WMMA GEMM: C = A[M,K](bf16) * Bt[N,K](bf16) ----------
// Double-buffered LDS filled by GLOBAL_LOAD_ASYNC_TO_LDS_B128 (ASYNCcnt).
// flags: 1=add bias(f32[N]), 2=add residual(f32[M,N]), 4=exact GELU, 8=bf16 out

#define GF_BIAS 1
#define GF_RES  2
#define GF_GELU 4
#define GF_OBF  8

__global__ __launch_bounds__(256)
void k_gemm_bf16(const __bf16* __restrict__ A, const __bf16* __restrict__ Bt,
                 int M, int N, int K,
                 const float* __restrict__ bias, const float* __restrict__ res,
                 void* __restrict__ outp, int flags) {
  __shared__ __bf16 As[2][128 * 40];
  __shared__ __bf16 Bs[2][128 * 40];
  const int tid  = threadIdx.x;
  const int lane = tid & 31, wave = tid >> 5;
  const int wm = wave & 3;          // 4 waves along M, 32 rows each
  const int wn = wave >> 2;         // 2 waves along N, 64 cols each
  const int bm = blockIdx.x * 128, bn = blockIdx.y * 128;
  const int lrow = tid >> 1, lcol = (tid & 1) << 4;   // 128 rows x 32 cols fill
  const int hi = lane >> 4, ln16 = lane & 15;

  v8f c[2][4] = {};

  const __bf16* Ag = A  + (size_t)(bm + lrow) * K + lcol;
  const __bf16* Bg = Bt + (size_t)(bn + lrow) * K + lcol;
  unsigned ldsA[2], ldsB[2];
  ldsA[0] = lds_off(&As[0][lrow * 40 + lcol]);
  ldsA[1] = lds_off(&As[1][lrow * 40 + lcol]);
  ldsB[0] = lds_off(&Bs[0][lrow * 40 + lcol]);
  ldsB[1] = lds_off(&Bs[1][lrow * 40 + lcol]);

  const int nk = K >> 5;
  // prologue: stage tile 0 into buffer 0
  async_b128(ldsA[0],      Ag);
  async_b128(ldsA[0] + 16, Ag + 8);
  async_b128(ldsB[0],      Bg);
  async_b128(ldsB[0] + 16, Bg + 8);
  Ag += 32; Bg += 32;

  for (int i = 0; i < nk; ++i) {
    const int cur = i & 1;
    if (i + 1 < nk) {
      const int nxt = cur ^ 1;
      async_b128(ldsA[nxt],      Ag);
      async_b128(ldsA[nxt] + 16, Ag + 8);
      async_b128(ldsB[nxt],      Bg);
      async_b128(ldsB[nxt] + 16, Bg + 8);
      Ag += 32; Bg += 32;
      __builtin_prefetch(Ag, 0, 1);
      __builtin_prefetch(Bg, 0, 1);
      WAIT_ASYNC(4);   // async ops complete in order: tile i has landed
    } else {
      WAIT_ASYNC(0);
    }
    __syncthreads();

    const __bf16* Ab = As[cur];
    const __bf16* Bb = Bs[cur];
    v16bf a0 = load_frag_a(Ab + (wm * 32     ) * 40, 40);
    v16bf a1 = load_frag_a(Ab + (wm * 32 + 16) * 40, 40);
    v16bf b0 = load_frag_b(Bb + (wn * 64     ) * 40, 40);
    v16bf b1 = load_frag_b(Bb + (wn * 64 + 16) * 40, 40);
    v16bf b2 = load_frag_b(Bb + (wn * 64 + 32) * 40, 40);
    v16bf b3 = load_frag_b(Bb + (wn * 64 + 48) * 40, 40);

    c[0][0] = wmma_bf16(a0, b0, c[0][0]);
    c[0][1] = wmma_bf16(a0, b1, c[0][1]);
    c[0][2] = wmma_bf16(a0, b2, c[0][2]);
    c[0][3] = wmma_bf16(a0, b3, c[0][3]);
    c[1][0] = wmma_bf16(a1, b0, c[1][0]);
    c[1][1] = wmma_bf16(a1, b1, c[1][1]);
    c[1][2] = wmma_bf16(a1, b2, c[1][2]);
    c[1][3] = wmma_bf16(a1, b3, c[1][3]);
    __syncthreads();   // all waves done reading buf[cur] before it is refilled
  }

  // epilogue: C layout = row (r + 8*hi) within tile, col = ln16
  for (int i = 0; i < 2; ++i)
    for (int j = 0; j < 4; ++j)
      for (int r = 0; r < 8; ++r) {
        const int mg = bm + wm * 32 + i * 16 + hi * 8 + r;
        const int ng = bn + wn * 64 + j * 16 + ln16;
        float val = c[i][j][r];
        if (flags & GF_BIAS) val += bias[ng];
        if (flags & GF_RES)  val += res[(size_t)mg * N + ng];
        if (flags & GF_GELU) val = 0.5f * val * (1.f + erff(val * 0.70710678118f));
        if (flags & GF_OBF)  ((__bf16*)outp)[(size_t)mg * N + ng] = f2bf(val);
        else                 ((float*) outp)[(size_t)mg * N + ng] = val;
      }
}

// ---------- flash attention: qkv bf16 [8192][3072] -> attn_out bf16 [8192][1024]
// grid = (32 q-tiles, 64 batch*head), 256 threads.  q-tile=64 rows, key-tile=64.

__global__ __launch_bounds__(256)
void k_attention(const __bf16* __restrict__ qkv, __bf16* __restrict__ attn_out) {
  __shared__ __bf16 qs [64 * 72];
  __shared__ __bf16 ks [64 * 72];
  __shared__ __bf16 vsT[64 * 72];   // transposed: [d][key]
  __shared__ float  ss [64 * 68];
  __shared__ __bf16 ps [64 * 72];
  __shared__ float  alpha_s[64];
  __shared__ float  l_s[64];

  const int tid = threadIdx.x, lane = tid & 31, wave = tid >> 5;
  const int wq = wave & 3;      // 4 waves along q (16 rows each)
  const int wx = wave >> 2;     // 2 waves along key/dh (32 cols each)
  const int hi = lane >> 4, ln16 = lane & 15;
  const int qt = blockIdx.x, bh = blockIdx.y;
  const int b = bh >> 4, h = bh & 15;
  const int rowQ = b * 2048 + qt * 64;
  const int trow = tid >> 2, tcol = (tid & 3) << 4;   // 64 rows x 64 cols loads

  {   // load Q tile (scaled at softmax time)
    const __bf16* src = qkv + (size_t)(rowQ + trow) * 3072 + h * 64 + tcol;
    __bf16* dst = qs + trow * 72 + tcol;
    *(uint4*)dst       = *(const uint4*)src;
    *(uint4*)(dst + 8) = *(const uint4*)(src + 8);
  }

  v8f o0 = {}, o1 = {};
  float m_run = -1e30f, l_run = 0.f;

  for (int kt = 0; kt < 32; ++kt) {
    const int rowK = b * 2048 + kt * 64;
    {   // K tile, row-major (this IS B^T layout for Q@K^T)
      const __bf16* src = qkv + (size_t)(rowK + trow) * 3072 + 1024 + h * 64 + tcol;
      __bf16* dst = ks + trow * 72 + tcol;
      *(uint4*)dst       = *(const uint4*)src;
      *(uint4*)(dst + 8) = *(const uint4*)(src + 8);
    }
    {   // V tile, transposed into LDS -> B^T layout for P@V
      const __bf16* src = qkv + (size_t)(rowK + trow) * 3072 + 2048 + h * 64 + tcol;
      union { uint4 u[2]; __bf16 e[16]; } t;
      t.u[0] = *(const uint4*)src;
      t.u[1] = *(const uint4*)(src + 8);
      for (int j = 0; j < 16; ++j) vsT[(tcol + j) * 72 + trow] = t.e[j];
    }
    __syncthreads();

    // S = Q @ K^T  (64x64, each wave a 16x32 strip)
    v8f s0 = {}, s1 = {};
    for (int kc = 0; kc < 2; ++kc) {
      v16bf aq = load_frag_a(qs + (wq * 16) * 72 + kc * 32, 72);
      v16bf k0 = load_frag_b(ks + (wx * 32     ) * 72 + kc * 32, 72);
      v16bf k1 = load_frag_b(ks + (wx * 32 + 16) * 72 + kc * 32, 72);
      s0 = wmma_bf16(aq, k0, s0);
      s1 = wmma_bf16(aq, k1, s1);
    }
    for (int r = 0; r < 8; ++r) {
      const int mr = wq * 16 + hi * 8 + r;
      ss[mr * 68 + wx * 32 + ln16]      = s0[r];
      ss[mr * 68 + wx * 32 + 16 + ln16] = s1[r];
    }
    __syncthreads();

    // online softmax: thread t < 64 owns q-row t
    if (tid < 64) {
      const float* sr = ss + tid * 68;
      float mx = -1e30f;
      for (int c2 = 0; c2 < 64; ++c2) mx = fmaxf(mx, sr[c2] * 0.125f);
      const float mnew = fmaxf(m_run, mx);
      const float al = __expf(m_run - mnew);
      float lsum = 0.f;
      for (int c2 = 0; c2 < 64; ++c2) {
        const float p = __expf(sr[c2] * 0.125f - mnew);
        ps[tid * 72 + c2] = f2bf(p);
        lsum += p;
      }
      l_run = l_run * al + lsum;
      m_run = mnew;
      alpha_s[tid] = al;
      l_s[tid] = l_run;
    }
    __syncthreads();

    // rescale running O, then O += P @ V
    float av[8];
    for (int r = 0; r < 8; ++r) av[r] = alpha_s[wq * 16 + hi * 8 + r];
    for (int r = 0; r < 8; ++r) { o0[r] *= av[r]; o1[r] *= av[r]; }

    for (int kc = 0; kc < 2; ++kc) {
      v16bf ap = load_frag_a(ps + (wq * 16) * 72 + kc * 32, 72);
      v16bf v0 = load_frag_b(vsT + (wx * 32     ) * 72 + kc * 32, 72);
      v16bf v1 = load_frag_b(vsT + (wx * 32 + 16) * 72 + kc * 32, 72);
      o0 = wmma_bf16(ap, v0, o0);
      o1 = wmma_bf16(ap, v1, o1);
    }
    __syncthreads();
  }

  // normalize by l and store bf16 into [8192][1024]
  for (int r = 0; r < 8; ++r) {
    const int mr = wq * 16 + hi * 8 + r;
    const float inv = 1.f / l_s[mr];
    const int grow = rowQ + mr;
    const int gcol = h * 64 + wx * 32 + ln16;
    attn_out[(size_t)grow * 1024 + gcol]      = f2bf(o0[r] * inv);
    attn_out[(size_t)grow * 1024 + gcol + 16] = f2bf(o1[r] * inv);
  }
}

// ---------- orchestration ----------

extern "C" void kernel_launch(void* const* d_in, const int* in_sizes, int n_in,
                              void* d_out, int out_size, void* d_ws, size_t ws_size,
                              hipStream_t stream) {
  (void)in_sizes; (void)n_in; (void)out_size; (void)ws_size;
  const float* x     = (const float*)d_in[0];
  const float* ln1_g = (const float*)d_in[1];
  const float* ln1_b = (const float*)d_in[2];
  const float* w_qkv = (const float*)d_in[3];   // [1024][3072]
  const float* w_out = (const float*)d_in[4];   // [1024][1024]
  const float* b_out = (const float*)d_in[5];
  const float* ln2_g = (const float*)d_in[6];
  const float* ln2_b = (const float*)d_in[7];
  const float* w1    = (const float*)d_in[8];   // [1024][4096]
  const float* b1    = (const float*)d_in[9];
  const float* w2    = (const float*)d_in[10];  // [4096][1024]
  const float* b2    = (const float*)d_in[11];
  float* out = (float*)d_out;

  const size_t M = 8192;
  unsigned char* w = (unsigned char*)d_ws;
  size_t o = 0;
  __bf16* xn1   = (__bf16*)(w + o); o += M * 1024 * 2;
  __bf16* wqkvT = (__bf16*)(w + o); o += (size_t)3072 * 1024 * 2;
  __bf16* qkv   = (__bf16*)(w + o); o += M * 3072 * 2;
  __bf16* attn  = (__bf16*)(w + o); o += M * 1024 * 2;
  __bf16* woutT = (__bf16*)(w + o); o += (size_t)1024 * 1024 * 2;
  float*  x1    = (float*) (w + o); o += M * 1024 * 4;
  __bf16* xn2   = (__bf16*)(w + o); o += M * 1024 * 2;
  __bf16* w1T   = (__bf16*)(w + o); o += (size_t)4096 * 1024 * 2;
  __bf16* hmid  = (__bf16*)(w + o); o += M * 4096 * 2;
  __bf16* w2T   = (__bf16*)(w + o); o += (size_t)1024 * 4096 * 2;

  // weight transpose + bf16 convert
  k_transpose_bf16<<<(1024 * 3072) / 256, 256, 0, stream>>>(w_qkv, wqkvT, 1024, 3072);
  k_transpose_bf16<<<(1024 * 1024) / 256, 256, 0, stream>>>(w_out, woutT, 1024, 1024);
  k_transpose_bf16<<<(1024 * 4096) / 256, 256, 0, stream>>>(w1, w1T, 1024, 4096);
  k_transpose_bf16<<<(4096 * 1024) / 256, 256, 0, stream>>>(w2, w2T, 4096, 1024);

  // attention sub-block
  k_layernorm<<<M, 256, 0, stream>>>(x, ln1_g, ln1_b, xn1);
  k_gemm_bf16<<<dim3(M / 128, 3072 / 128), 256, 0, stream>>>(
      xn1, wqkvT, M, 3072, 1024, nullptr, nullptr, qkv, GF_OBF);
  k_attention<<<dim3(32, 64), 256, 0, stream>>>(qkv, attn);
  k_gemm_bf16<<<dim3(M / 128, 1024 / 128), 256, 0, stream>>>(
      attn, woutT, M, 1024, 1024, b_out, x, x1, GF_BIAS | GF_RES);

  // MLP sub-block
  k_layernorm<<<M, 256, 0, stream>>>(x1, ln2_g, ln2_b, xn2);
  k_gemm_bf16<<<dim3(M / 128, 4096 / 128), 256, 0, stream>>>(
      xn2, w1T, M, 4096, 1024, b1, nullptr, hmid, GF_BIAS | GF_GELU | GF_OBF);
  k_gemm_bf16<<<dim3(M / 128, 1024 / 128), 256, 0, stream>>>(
      hmid, w2T, M, 1024, 4096, b2, x1, out, GF_BIAS | GF_RES);
}
